// DHMSA_32976758899375
// MI455X (gfx1250) — matrix operand order repaired and checked by reading, not compiled, not run on hardware
//
#include <hip/hip_runtime.h>
#include <hip/hip_bf16.h>

typedef __attribute__((ext_vector_type(16))) _Float16 v16h;
typedef __attribute__((ext_vector_type(8)))  float    v8f;

// ---------------------------------------------------------------------------
// WMMA fragment loaders per CDNA5 ISA §7.12.2 (wave32, 16-bit A, f16 B, f32 C/D)
// A 16x32 (MxK): lane m = lane&15 (both halves), VGPR v: K = (v>=4?16:0)+(half?8:0)+2*(v&3)
// B 32x16 (KxN): lane n = lane&15 (both halves), VGPR v: K = (v>=4?16:0)+(half?8:0)+2*(v&3) (same as A)
//   -> B staged TRANSPOSED in LDS ([N][K]) so each pair is a contiguous 32-bit read.
// C/D 16x16: VGPR r: M = r + 8*half, N = lane&15
// ---------------------------------------------------------------------------
__device__ __forceinline__ v16h frag_a16(const _Float16* p, int row0, int stride,
                                         int kbase, int lane) {
  const int m  = lane & 15;
  const int hf = (lane >> 4) & 1;
  union { v16h v; _Float16 h[16]; } f;
#pragma unroll
  for (int r = 0; r < 8; ++r) {
    const int k0 = kbase + ((r & 4) << 2) + (hf << 3) + ((r & 3) << 1);
    const _Float16* q = p + (row0 + m) * stride + k0;
    f.h[2 * r]     = q[0];
    f.h[2 * r + 1] = q[1];
  }
  return f.v;
}

__device__ __forceinline__ v16h frag_b16(const _Float16* pT, int col0, int stride,
                                         int kbase, int lane) {
  const int n  = lane & 15;
  const int hf = (lane >> 4) & 1;
  union { v16h v; _Float16 h[16]; } f;
#pragma unroll
  for (int r = 0; r < 8; ++r) {
    const int k0 = kbase + ((r & 4) << 2) + (hf << 3) + ((r & 3) << 1);
    const _Float16* q = pT + (col0 + n) * stride + k0;
    f.h[2 * r]     = q[0];
    f.h[2 * r + 1] = q[1];
  }
  return f.v;
}

union H4 { _Float16 h[4]; unsigned long long u; };

// ---------------------------------------------------------------------------
// Generic WMMA GEMM: C[MxN] = A[MxK] * B[KxN], fp32 in/out, f16 WMMA compute.
// N, K compile-time -> constant strides, immediate store offsets.
// Block tile 128x64, 8 waves (4x2), each wave 32x32 (2x2 WMMA tiles), K step 32.
// ---------------------------------------------------------------------------
#define GA_STRIDE 40
#define GB_STRIDE 40

template <int N, int K>
__global__ __launch_bounds__(256) void gemm_wmma_f16(
    const float* __restrict__ A, const float* __restrict__ B,
    float* __restrict__ C)
{
  __shared__ __align__(16) _Float16 lA[128 * GA_STRIDE];   // A tile [128][32]
  __shared__ __align__(16) _Float16 lBt[64 * GB_STRIDE];   // B tile transposed [64][32]

  const int t    = threadIdx.x;
  const int lane = t & 31;
  const int wave = t >> 5;
  const int wm   = wave & 3;      // 0..3 -> rows
  const int wn   = wave >> 2;     // 0..1 -> cols
  const int mB   = blockIdx.x * 128;
  const int nB   = blockIdx.y * 64;

  v8f zero = {};
  v8f acc[2][2];
  acc[0][0] = zero; acc[0][1] = zero; acc[1][0] = zero; acc[1][1] = zero;

  for (int kb = 0; kb < K; kb += 32) {
    __syncthreads();
    // A tile: 128x32 floats = 1024 float4; 4 per thread (global_load_b128)
#pragma unroll
    for (int i = 0; i < 4; ++i) {
      const int c4 = t + i * 256;            // float4 chunk id
      const int r  = c4 >> 3;                // 8 chunks per 32-col row
      const int co = (c4 & 7) * 4;
      const float4 f = *(const float4*)(A + (size_t)(mB + r) * K + kb + co);
      H4 p;
      p.h[0] = (_Float16)f.x; p.h[1] = (_Float16)f.y;
      p.h[2] = (_Float16)f.z; p.h[3] = (_Float16)f.w;
      *(unsigned long long*)(&lA[r * GA_STRIDE + co]) = p.u;
    }
    // B tile: 32x64 floats = 512 float4; 2 per thread, scatter transposed
#pragma unroll
    for (int i = 0; i < 2; ++i) {
      const int c4 = t + i * 256;
      const int kk = c4 >> 4;                // 16 chunks per 64-col row
      const int nn = (c4 & 15) * 4;
      const float4 f = *(const float4*)(B + (size_t)(kb + kk) * N + nB + nn);
      lBt[(nn + 0) * GB_STRIDE + kk] = (_Float16)f.x;
      lBt[(nn + 1) * GB_STRIDE + kk] = (_Float16)f.y;
      lBt[(nn + 2) * GB_STRIDE + kk] = (_Float16)f.z;
      lBt[(nn + 3) * GB_STRIDE + kk] = (_Float16)f.w;
    }
    __syncthreads();

    v16h bf[2];
#pragma unroll
    for (int sn = 0; sn < 2; ++sn)
      bf[sn] = frag_b16(lBt, wn * 32 + sn * 16, GB_STRIDE, 0, lane);
#pragma unroll
    for (int sm = 0; sm < 2; ++sm) {
      v16h af = frag_a16(lA, wm * 32 + sm * 16, GA_STRIDE, 0, lane);
#pragma unroll
      for (int sn = 0; sn < 2; ++sn)
        acc[sm][sn] = __builtin_amdgcn_wmma_f32_16x16x32_f16(
            false, af, false, bf[sn], (short)0, acc[sm][sn], false, false);
    }
  }

  const int nlo = lane & 15, hf = (lane >> 4) & 1;
  float* Cb = C + (size_t)(mB + wm * 32 + 8 * hf) * N + nB + wn * 32 + nlo;
#pragma unroll
  for (int sm = 0; sm < 2; ++sm)
#pragma unroll
    for (int sn = 0; sn < 2; ++sn)
#pragma unroll
      for (int r = 0; r < 8; ++r)
        Cb[(size_t)(sm * 16 + r) * N + sn * 16] = acc[sm][sn][r];
}

// ---------------------------------------------------------------------------
// Depthwise 3x3 conv (SAME, per-channel) + LayerNorm(768) + q/v bias.
// One block per pixel; thread t owns channels {t, t+256, t+512}.
// ---------------------------------------------------------------------------
__global__ __launch_bounds__(256) void dwconv_ln(
    const float* __restrict__ qkv, const float* __restrict__ w_dw,
    const float* __restrict__ ln_g, const float* __restrict__ ln_b,
    const float* __restrict__ q_bias, const float* __restrict__ v_bias,
    float* __restrict__ out)
{
  __shared__ float rS[256];
  __shared__ float rS2[256];
  const int pix = blockIdx.x;            // 0..32767
  const int b = pix >> 14, y = (pix >> 7) & 127, x = pix & 127;
  const int t = threadIdx.x;
  const size_t pbase = (size_t)pix * 768;

  float v[3], s = 0.f, s2 = 0.f;
#pragma unroll
  for (int cc = 0; cc < 3; ++cc) {
    const int ch = t + cc * 256;
    float a = 0.f;
#pragma unroll
    for (int dy = -1; dy <= 1; ++dy) {
      const int yy = y + dy;
      if (yy < 0 || yy > 127) continue;
#pragma unroll
      for (int dx = -1; dx <= 1; ++dx) {
        const int xx = x + dx;
        if (xx < 0 || xx > 127) continue;
        a += qkv[((size_t)((b * 128 + yy) * 128 + xx)) * 768 + ch] *
             w_dw[((dy + 1) * 3 + (dx + 1)) * 768 + ch];
      }
    }
    v[cc] = a; s += a; s2 += a * a;
  }
  rS[t] = s; rS2[t] = s2;
  __syncthreads();
  for (int off = 128; off > 0; off >>= 1) {
    if (t < off) { rS[t] += rS[t + off]; rS2[t] += rS2[t + off]; }
    __syncthreads();
  }
  const float mu   = rS[0] * (1.f / 768.f);
  const float var  = rS2[0] * (1.f / 768.f) - mu * mu;
  const float rstd = rsqrtf(var + 1e-5f);
#pragma unroll
  for (int cc = 0; cc < 3; ++cc) {
    const int ch = t + cc * 256;
    const float bias = (ch < 256) ? q_bias[ch] : ((ch < 512) ? 0.f : v_bias[ch - 512]);
    out[pbase + ch] = (v[cc] - mu) * rstd * ln_g[ch] + ln_b[ch] + bias;
  }
}

// ---------------------------------------------------------------------------
// CPB MLP: bias_tab[529][8]. _REL_TAB coordinates recomputed on device.
// ---------------------------------------------------------------------------
__device__ __forceinline__ float rel_coord(int j) {
  const float c = (float)(j - 11) * (8.f / 7.f);
  const float sgn = (c > 0.f) ? 1.f : ((c < 0.f) ? -1.f : 0.f);
  return sgn * log1pf(fabsf(c)) * (1.f / 2.0794415416798357f);   // /ln(8)
}

__global__ __launch_bounds__(256) void cpb_mlp(
    const float* __restrict__ w1, const float* __restrict__ b1,
    const float* __restrict__ w2, float* __restrict__ bias_tab)
{
  __shared__ float hid[512];
  const int row = blockIdx.x;                 // 0..528
  const float v0 = rel_coord(row / 23);
  const float v1 = rel_coord(row % 23);
  const int t = threadIdx.x;
#pragma unroll
  for (int i = 0; i < 2; ++i) {
    const int hh = t + i * 256;
    const float a = v0 * w1[hh] + v1 * w1[512 + hh] + b1[hh];
    hid[hh] = fmaxf(a, 0.f);
  }
  __syncthreads();
  if (t < 8) {
    float sum = 0.f;
    for (int hh = 0; hh < 512; ++hh) sum += hid[hh] * w2[hh * 8 + t];
    bias_tab[row * 8 + t] = 16.f / (1.f + __expf(-sum));
  }
}

// rb[h][q][k] = bias_tab[(qy-ky+15)*23 + (qx-kx+15)][h]
__global__ __launch_bounds__(256) void rb_expand(
    const float* __restrict__ bias_tab, float* __restrict__ rb)
{
  const int hq = blockIdx.x;                  // 0..511
  const int h = hq >> 6, q = hq & 63;
  const int k = threadIdx.x;                  // 0..255
  const int qy = q >> 3, qx = q & 7, ky = k >> 4, kx = k & 15;
  const int idx = (qy - ky + 15) * 23 + (qx - kx + 15);
  rb[(size_t)hq * 256 + k] = bias_tab[idx * 8 + h];
}

// ---------------------------------------------------------------------------
// Fused halo-window attention. One block per (window, head); 8 waves.
//  phase1: l2n(q)*scale, l2n(k) -> f16 LDS (zero-pad halo), float4 global loads
//  phase2: S = q k^T via WMMA (8 waves x 8 tiles), scores -> f16 LDS
//  phase3: reload v transposed into dead q/k LDS; softmax(S + rb), 4 thr/row
//  phase4: O = P v via WMMA, scatter to NHWC out buffer
// Static LDS: 25600 + 33792 + 1024 = 60416 B.
// ---------------------------------------------------------------------------
__global__ __launch_bounds__(256) void attn_kernel(
    const float* __restrict__ qkvn, const float* __restrict__ rb,
    const float* __restrict__ logit_scale, float* __restrict__ outb)
{
  __shared__ __align__(16) _Float16 smA[12800]; // qs[64*40]+ks[256*40]; later vt[32*264]
  __shared__ __align__(16) _Float16 sc[64 * 264]; // scores / probs (f16)
  __shared__ float red[256];                    // softmax cross-thread combine
  _Float16* qs = smA;                    // [64][40]
  _Float16* ks = smA + 64 * 40;          // [256][40]  (B-transposed for q k^T)
  _Float16* vt = smA;                    // [32][264]  (v transposed, for P v)

  const int w = blockIdx.x >> 3;         // 0..511
  const int h = blockIdx.x & 7;
  const int b = w >> 8;
  const int wy = (w >> 4) & 15, wx = w & 15;
  const int t = threadIdx.x;
  const int lane = t & 31;
  const int wave = t >> 5;

  // ---- phase 1: normalized k (all 256 threads), normalized+scaled q (t<64)
  {
    const int ky = t >> 4, kx = t & 15;
    const int gy = wy * 8 - 4 + ky, gx = wx * 8 - 4 + kx;
    const bool ok = (gy >= 0) & (gy < 128) & (gx >= 0) & (gx < 128);
    const float* px = qkvn + ((size_t)((b * 128 + gy) * 128 + gx)) * 768 + 256 + h * 32;
    float4 kv[8];
#pragma unroll
    for (int i = 0; i < 8; ++i)
      kv[i] = ok ? *(const float4*)(px + i * 4) : float4{0.f, 0.f, 0.f, 0.f};
    float ss = 0.f;
#pragma unroll
    for (int i = 0; i < 8; ++i)
      ss += kv[i].x * kv[i].x + kv[i].y * kv[i].y + kv[i].z * kv[i].z + kv[i].w * kv[i].w;
    const float rn = rsqrtf(fmaxf(ss, 1.55e-5f));
#pragma unroll
    for (int i = 0; i < 8; ++i) {
      H4 p;
      p.h[0] = (_Float16)(kv[i].x * rn); p.h[1] = (_Float16)(kv[i].y * rn);
      p.h[2] = (_Float16)(kv[i].z * rn); p.h[3] = (_Float16)(kv[i].w * rn);
      *(unsigned long long*)(&ks[t * 40 + i * 4]) = p.u;
    }
  }
  if (t < 64) {
    const float lsv = logit_scale[h];
    const float scale = __expf(fminf(lsv, 4.6051702f));     // ln(100)
    const int qy = t >> 3, qx = t & 7;
    const int gy = wy * 8 + qy, gx = wx * 8 + qx;
    const float* px = qkvn + ((size_t)((b * 128 + gy) * 128 + gx)) * 768 + h * 32;
    float4 qv[8];
    float ss = 0.f;
#pragma unroll
    for (int i = 0; i < 8; ++i) {
      qv[i] = *(const float4*)(px + i * 4);
      ss += qv[i].x * qv[i].x + qv[i].y * qv[i].y + qv[i].z * qv[i].z + qv[i].w * qv[i].w;
    }
    const float rn = rsqrtf(fmaxf(ss, 1.55e-5f)) * scale;
#pragma unroll
    for (int i = 0; i < 8; ++i) {
      H4 p;
      p.h[0] = (_Float16)(qv[i].x * rn); p.h[1] = (_Float16)(qv[i].y * rn);
      p.h[2] = (_Float16)(qv[i].z * rn); p.h[3] = (_Float16)(qv[i].w * rn);
      *(unsigned long long*)(&qs[t * 40 + i * 4]) = p.u;
    }
  }
  __syncthreads();

  // ---- phase 2: S(64x256) = q(64x32) k^T ; wave -> n stripe [wave*32,+32)
  {
    const int nB = wave * 32;
    v8f zero = {};
    v16h bf0 = frag_b16(ks, nB,      40, 0, lane);
    v16h bf1 = frag_b16(ks, nB + 16, 40, 0, lane);
    const int nlo = lane & 15, hf = (lane >> 4) & 1;
#pragma unroll
    for (int mt = 0; mt < 4; ++mt) {
      v16h af = frag_a16(qs, mt * 16, 40, 0, lane);
      v8f c0 = __builtin_amdgcn_wmma_f32_16x16x32_f16(false, af, false, bf0,
                                                      (short)0, zero, false, false);
      v8f c1 = __builtin_amdgcn_wmma_f32_16x16x32_f16(false, af, false, bf1,
                                                      (short)0, zero, false, false);
#pragma unroll
      for (int r = 0; r < 8; ++r) {
        const int m = mt * 16 + r + 8 * hf;
        sc[m * 264 + nB + nlo]      = (_Float16)c0[r];
        sc[m * 264 + nB + 16 + nlo] = (_Float16)c1[r];
      }
    }
  }
  __syncthreads();

  // ---- phase 3a: reload v transposed into smA (q/k regions now dead)
  {
    const int ky = t >> 4, kx = t & 15;
    const int gy = wy * 8 - 4 + ky, gx = wx * 8 - 4 + kx;
    const bool ok = (gy >= 0) & (gy < 128) & (gx >= 0) & (gx < 128);
    const float* px = qkvn + ((size_t)((b * 128 + gy) * 128 + gx)) * 768 + 512 + h * 32;
#pragma unroll
    for (int i = 0; i < 8; ++i) {
      const float4 f = ok ? *(const float4*)(px + i * 4) : float4{0.f, 0.f, 0.f, 0.f};
      vt[(i * 4 + 0) * 264 + t] = (_Float16)f.x;
      vt[(i * 4 + 1) * 264 + t] = (_Float16)f.y;
      vt[(i * 4 + 2) * 264 + t] = (_Float16)f.z;
      vt[(i * 4 + 3) * 264 + t] = (_Float16)f.w;
    }
  }
  // ---- phase 3b: softmax(S + rb); 4 threads per row, LDS max/sum combine
  {
    const int row = t >> 2, seg = t & 3;
    const float* rbp = rb + ((size_t)(h * 64 + row)) * 256 + seg * 64;
    _Float16* scp = sc + row * 264 + seg * 64;
    float mx = -3.4e38f;
    for (int j = 0; j < 64; ++j) mx = fmaxf(mx, (float)scp[j] + rbp[j]);
    red[t] = mx;
    __syncthreads();
    mx = fmaxf(fmaxf(red[row * 4 + 0], red[row * 4 + 1]),
               fmaxf(red[row * 4 + 2], red[row * 4 + 3]));
    __syncthreads();                              // protect red before reuse
    float sum = 0.f;
    for (int j = 0; j < 64; ++j) {
      const float e = __expf((float)scp[j] + rbp[j] - mx);
      sum += e;
      scp[j] = (_Float16)e;
    }
    red[t] = sum;
    __syncthreads();
    sum = (red[row * 4 + 0] + red[row * 4 + 1]) +
          (red[row * 4 + 2] + red[row * 4 + 3]);
    const float inv = 1.f / sum;
    for (int j = 0; j < 64; ++j)
      scp[j] = (_Float16)((float)scp[j] * inv);
  }
  __syncthreads();

  // ---- phase 4: O(64x32) = P(64x256) v(256x32); wave -> tile (mt, nt)
  {
    const int mt = wave >> 1, nt = wave & 1;
    v8f acc = {};
#pragma unroll
    for (int kc = 0; kc < 8; ++kc) {
      v16h af = frag_a16(sc, mt * 16, 264, kc * 32, lane);
      v16h bf = frag_b16(vt, nt * 16, 264, kc * 32, lane);
      acc = __builtin_amdgcn_wmma_f32_16x16x32_f16(false, af, false, bf,
                                                   (short)0, acc, false, false);
    }
    const int nlo = lane & 15, hf = (lane >> 4) & 1;
#pragma unroll
    for (int r = 0; r < 8; ++r) {
      const int m = mt * 16 + r + 8 * hf;        // query index 0..63
      const int n = nt * 16 + nlo;               // head-dim 0..31
      const int gy = wy * 8 + (m >> 3), gx = wx * 8 + (m & 7);
      outb[((size_t)((b * 128 + gy) * 128 + gx)) * 256 + h * 32 + n] = acc[r];
    }
  }
}

// ---------------------------------------------------------------------------
// Launch: qkv GEMM -> dwconv+LN -> CPB/rb -> attention -> output projection
// Workspace (floats): qkv_raw 25165824 | qkv_norm 25165824 | attn_out 8388608
//                     | bias_tab 4232 | rb 131072   (~235.4 MB total)
// ---------------------------------------------------------------------------
extern "C" void kernel_launch(void* const* d_in, const int* in_sizes, int n_in,
                              void* d_out, int out_size, void* d_ws, size_t ws_size,
                              hipStream_t stream) {
  const float* x           = (const float*)d_in[0];
  const float* w_qkv       = (const float*)d_in[1];
  const float* w_dw        = (const float*)d_in[2];
  const float* ln_g        = (const float*)d_in[3];
  const float* ln_b        = (const float*)d_in[4];
  const float* q_bias      = (const float*)d_in[5];
  const float* v_bias      = (const float*)d_in[6];
  const float* logit_scale = (const float*)d_in[7];
  const float* cpb_w1      = (const float*)d_in[8];
  const float* cpb_b1      = (const float*)d_in[9];
  const float* cpb_w2      = (const float*)d_in[10];
  const float* w_proj      = (const float*)d_in[11];
  float* out = (float*)d_out;

  float* ws       = (float*)d_ws;
  float* qkv_raw  = ws;
  float* qkv_norm = qkv_raw  + 25165824ull;
  float* attn_out = qkv_norm + 25165824ull;
  float* bias_tab = attn_out + 8388608ull;
  float* rb       = bias_tab + 4232ull;

  // 1) QKV projection: (32768x256)*(256x768)
  gemm_wmma_f16<768, 256><<<dim3(256, 12), 256, 0, stream>>>(x, w_qkv, qkv_raw);
  // 2) depthwise conv + LN + bias
  dwconv_ln<<<32768, 256, 0, stream>>>(qkv_raw, w_dw, ln_g, ln_b, q_bias, v_bias, qkv_norm);
  // 3) CPB MLP + rb table
  cpb_mlp<<<529, 256, 0, stream>>>(cpb_w1, cpb_b1, cpb_w2, bias_tab);
  rb_expand<<<512, 256, 0, stream>>>(bias_tab, rb);
  // 4) halo-window attention
  attn_kernel<<<4096, 256, 0, stream>>>(qkv_norm, rb, logit_scale, attn_out);
  // 5) output projection: (32768x256)*(256x256)
  gemm_wmma_f16<256, 256><<<dim3(256, 4), 256, 0, stream>>>(attn_out, w_proj, out);
}
